// Quantizer_14920716386844
// MI455X (gfx1250) — matrix operand order repaired and checked by reading, not compiled, not run on hardware
//
#include <hip/hip_runtime.h>
#include <hip/hip_bf16.h>

// ---------------------------------------------------------------------------
// VQ forward for MI455X (gfx1250, wave32, WMMA).
//   inputs  : [16, 256, 64, 64] f32
//   codebook: [1024, 256] f32
//   out     : quantized [16,256,64,64] f32  ++  commit_loss scalar (1 float)
// Distance GEMM (65536x1024x256) done with v_wmma_f32_16x16x32_bf16 using a
// 2-term bf16 hi/lo split (3 WMMAs per tile) for ~f32 dot-product accuracy,
// fused with per-row argmin. Gather uses the original f32 codebook (exact).
// ---------------------------------------------------------------------------

#define EMBED_DIM 256
#define CODEBOOK_SIZE 1024
#define BATCH 16
#define HW 4096              // 64*64
#define ROWS_PER_WG 64
#define NUM_WG 1024          // 16*4096/64
#define PITCH 264            // bf16 halves per LDS row (256 + 8 pad)
#define NUMEL_INV 5.9604644775390625e-8f  // 1/(16*256*4096)

typedef __bf16  v16bf __attribute__((ext_vector_type(16)));
typedef __bf16  bf16x8 __attribute__((ext_vector_type(8)));
typedef float   v8f   __attribute__((ext_vector_type(8)));

// A fragment (16x32 bf16, row-major x): lane covers row m, khalf hi=lane>>4.
// halves j=0..7  -> K = 8*hi + j        (contiguous 16B)
// halves j=8..15 -> K = 16 + 8*hi + j-8 (contiguous 16B)
__device__ inline v16bf load_frag_a(const __bf16* p) {
  bf16x8 a0 = *(const bf16x8*)(p);
  bf16x8 a1 = *(const bf16x8*)(p + 16);
  v16bf r;
#pragma unroll
  for (int j = 0; j < 8; ++j) { r[j] = a0[j]; r[8 + j] = a1[j]; }
  return r;
}

// B fragment (32x16 bf16, codebook^T): lane covers code n=lane&15, kg=lane>>4.
// halves j=0..7 -> K = 8*kg + j ; j=8..15 -> K = 16 + 8*kg + j-8.
__device__ inline v16bf load_frag_b(const __bf16* p) {
  bf16x8 b0 = *(const bf16x8*)(p);
  bf16x8 b1 = *(const bf16x8*)(p + 16);
  v16bf r;
#pragma unroll
  for (int j = 0; j < 8; ++j) { r[j] = b0[j]; r[8 + j] = b1[j]; }
  return r;
}

// ---------------------------------------------------------------------------
// Kernel 1: codebook -> bf16 hi/lo split + ||c||^2 (deterministic reduce)
// grid = 1024 blocks x 256 threads (one block per code)
// ---------------------------------------------------------------------------
__global__ void vq_prep_codebook(const float* __restrict__ cb,
                                 __bf16* __restrict__ cbh,
                                 __bf16* __restrict__ cbl,
                                 float* __restrict__ c2) {
  __shared__ float red[256];
  const int k = blockIdx.x;
  const int t = threadIdx.x;
  const float v = cb[(size_t)k * EMBED_DIM + t];
  const __bf16 h  = (__bf16)v;
  const __bf16 lo = (__bf16)(v - (float)h);
  cbh[(size_t)k * EMBED_DIM + t] = h;
  cbl[(size_t)k * EMBED_DIM + t] = lo;
  red[t] = v * v;
  __syncthreads();
#pragma unroll
  for (int s = 128; s > 0; s >>= 1) {
    if (t < s) red[t] += red[t + s];
    __syncthreads();
  }
  if (t == 0) c2[k] = red[0];
}

// ---------------------------------------------------------------------------
// Kernel 2: fused distance-GEMM + argmin + gather + loss partial
// grid = 1024 WGs x 256 threads (8 waves of 32). WG owns 64 consecutive
// pixels of one image. Waves: rg = wave&3 (16-row group), cs = wave>>2
// (code stripe: codes kt = cs*16, cs*16+32, ...).
// ---------------------------------------------------------------------------
__global__ void __launch_bounds__(256, 2)
vq_main(const float* __restrict__ x,
        const float* __restrict__ cb,
        const __bf16* __restrict__ cbh,
        const __bf16* __restrict__ cbl,
        const float* __restrict__ c2,
        float* __restrict__ out,
        float* __restrict__ loss_partial) {
  extern __shared__ char smem[];
  __bf16* xh  = (__bf16*)smem;                       // 64*264 halves
  __bf16* xl  = xh + ROWS_PER_WG * PITCH;            // 64*264 halves
  float*  x2s = (float*)(xl + ROWS_PER_WG * PITCH);  // 64
  float*  bestS = x2s + 64;                          // 128 (2 stripes x 64)
  int*    idxS  = (int*)(bestS + 128);               // 128
  int*    fidx  = idxS + 128;                        // 64
  float*  red   = (float*)(fidx + 64);               // 256

  const int t   = threadIdx.x;
  const int b   = blockIdx.x >> 6;            // image
  const int hw0 = (blockIdx.x & 63) * 64;     // first pixel
  const int m   = t & 63;                     // pixel within tile
  const int c4  = t >> 6;                     // 0..3

  // ---- stage x (transpose + bf16 split), accumulate ||x||^2 partials ----
  const float* xg = x + ((size_t)b * EMBED_DIM) * HW + hw0 + m;
  float sq = 0.f;
#pragma unroll 4
  for (int i = 0; i < 64; ++i) {
    const int c = c4 + 4 * i;
    const float v = xg[(size_t)c * HW];
    const __bf16 h  = (__bf16)v;
    const __bf16 lo = (__bf16)(v - (float)h);
    xh[m * PITCH + c] = h;
    xl[m * PITCH + c] = lo;
    sq += v * v;
  }
  red[c4 * 64 + m] = sq;
  __syncthreads();
  if (t < 64) x2s[t] = (red[t] + red[64 + t]) + (red[128 + t] + red[192 + t]);
  __syncthreads();

  // ---- WMMA distance GEMM with fused running argmin ----
  const int w  = t >> 5;      // wave 0..7
  const int l  = t & 31;      // lane
  const int rg = w & 3;       // row group (16 rows)
  const int cs = w >> 2;      // code stripe (0/1)
  const int ln = l & 15;
  const int hi = l >> 4;

  float x2v[8];
#pragma unroll
  for (int v = 0; v < 8; ++v) x2v[v] = x2s[16 * rg + hi * 8 + v];

  float best[8];
  int   bidx[8];
#pragma unroll
  for (int v = 0; v < 8; ++v) { best[v] = 3.4e38f; bidx[v] = 0; }

  const __bf16* arow_h = xh + (16 * rg + ln) * PITCH + 8 * hi;
  const __bf16* arow_l = xl + (16 * rg + ln) * PITCH + 8 * hi;

  for (int kt = cs * 16; kt < CODEBOOK_SIZE; kt += 32) {
    const int code = kt + ln;
    const __bf16* bh_p = cbh + (size_t)code * EMBED_DIM + 8 * hi;
    const __bf16* bl_p = cbl + (size_t)code * EMBED_DIM + 8 * hi;

    v8f acc = {};
#pragma unroll
    for (int cc = 0; cc < 8; ++cc) {
      const int c0 = cc * 32;
      const v16bf ah = load_frag_a(arow_h + c0);
      const v16bf al = load_frag_a(arow_l + c0);
      const v16bf bh = load_frag_b(bh_p + c0);
      const v16bf bl = load_frag_b(bl_p + c0);
      acc = __builtin_amdgcn_wmma_f32_16x16x32_bf16(false, ah, false, bh,
                                                    (short)0, acc, false, false);
      acc = __builtin_amdgcn_wmma_f32_16x16x32_bf16(false, ah, false, bl,
                                                    (short)0, acc, false, false);
      acc = __builtin_amdgcn_wmma_f32_16x16x32_bf16(false, al, false, bh,
                                                    (short)0, acc, false, false);
    }

    const float c2v = c2[code];
#pragma unroll
    for (int v = 0; v < 8; ++v) {
      const float d = (x2v[v] + c2v) - 2.0f * acc[v];
      if (d < best[v]) { best[v] = d; bidx[v] = code; }
    }
  }

  // ---- cross-lane argmin (16-lane butterflies; halves stay separate) ----
#pragma unroll
  for (int off = 1; off < 16; off <<= 1) {
#pragma unroll
    for (int v = 0; v < 8; ++v) {
      const float ob = __shfl_xor(best[v], off, 32);
      const int   oi = __shfl_xor(bidx[v], off, 32);
      if (ob < best[v] || (ob == best[v] && oi < bidx[v])) {
        best[v] = ob; bidx[v] = oi;
      }
    }
  }
  if (ln == 0) {
#pragma unroll
    for (int v = 0; v < 8; ++v) {
      const int r = 16 * rg + hi * 8 + v;
      bestS[cs * 64 + r] = best[v];
      idxS [cs * 64 + r] = bidx[v];
    }
  }
  __syncthreads();
  if (t < 64) {
    const float b0 = bestS[t], b1 = bestS[64 + t];
    const int   i0 = idxS[t],  i1 = idxS[64 + t];
    fidx[t] = (b1 < b0 || (b1 == b0 && i1 < i0)) ? i1 : i0;
  }
  __syncthreads();

  // ---- gather (exact f32 codebook), write output, loss partial ----
  const int k = fidx[m];
  const float* cbrow = cb + (size_t)k * EMBED_DIM;
  float* og = out + ((size_t)b * EMBED_DIM) * HW + hw0 + m;
  float lsum = 0.f;
#pragma unroll 4
  for (int i = 0; i < 64; ++i) {
    const int c = c4 + 4 * i;
    const float q  = cbrow[c];
    const float xv = (float)xh[m * PITCH + c] + (float)xl[m * PITCH + c];
    og[(size_t)c * HW] = q;
    const float d = q - xv;
    lsum += d * d;
  }
  red[t] = lsum;
  __syncthreads();
#pragma unroll
  for (int s = 128; s > 0; s >>= 1) {
    if (t < s) red[t] += red[t + s];
    __syncthreads();
  }
  if (t == 0) loss_partial[blockIdx.x] = red[0];
}

// ---------------------------------------------------------------------------
// Kernel 3: deterministic fold of 1024 partials -> commit loss scalar
// ---------------------------------------------------------------------------
__global__ void vq_loss_reduce(const float* __restrict__ partial,
                               float* __restrict__ loss_out) {
  __shared__ float red[256];
  const int t = threadIdx.x;
  red[t] = (partial[t] + partial[t + 256]) + (partial[t + 512] + partial[t + 768]);
  __syncthreads();
#pragma unroll
  for (int s = 128; s > 0; s >>= 1) {
    if (t < s) red[t] += red[t + s];
    __syncthreads();
  }
  if (t == 0) loss_out[0] = red[0] * NUMEL_INV;
}

// ---------------------------------------------------------------------------
extern "C" void kernel_launch(void* const* d_in, const int* in_sizes, int n_in,
                              void* d_out, int out_size, void* d_ws, size_t ws_size,
                              hipStream_t stream) {
  const float* x  = (const float*)d_in[0];   // [16,256,64,64]
  const float* cb = (const float*)d_in[1];   // [1024,256]
  float* out = (float*)d_out;                // quantized ++ loss

  char* ws = (char*)d_ws;
  __bf16* cbh = (__bf16*)ws;                                   // 512 KiB
  __bf16* cbl = (__bf16*)(ws + 524288);                        // 512 KiB
  float*  c2  = (float*)(ws + 1048576);                        // 4 KiB
  float*  part = (float*)(ws + 1048576 + 4096);                // 4 KiB

  const size_t out_elems = (size_t)BATCH * EMBED_DIM * HW;     // 16777216
  float* loss_out = out + out_elems;

  vq_prep_codebook<<<CODEBOOK_SIZE, 256, 0, stream>>>(cb, cbh, cbl, c2);

  const size_t lds_bytes =
      (size_t)2 * ROWS_PER_WG * PITCH * sizeof(__bf16)   // xh + xl
      + 64 * sizeof(float)                               // x2s
      + 128 * sizeof(float) + 128 * sizeof(int)          // bestS + idxS
      + 64 * sizeof(int)                                 // fidx
      + 256 * sizeof(float);                             // red
  vq_main<<<NUM_WG, 256, lds_bytes, stream>>>(x, cb, cbh, cbl, c2, out, part);

  vq_loss_reduce<<<1, 256, 0, stream>>>(part, loss_out);
}
